// LlamaAttentionFused_87265145520305
// MI455X (gfx1250) — compile-verified, matrix operand order checked by reading, not run on hardware
//
#include <hip/hip_runtime.h>
#include <cstdint>
#include <cstddef>

// ---------------------------------------------------------------------------
// Llama attention block, fused for MI455X (gfx1250, wave32, WMMA).
// Pipeline: cvt(x,W)->bf16 ; GEMM(QKV) ; RoPE+split ; flash-attention ; GEMM(out)
// Matrix math: v_wmma_f32_16x16x32_bf16 (f32 accumulate).
// Data movement:
//   * GEMM tiles: Tensor Data Mover (tensor_load_to_lds + s_wait_tensorcnt),
//     double-buffered, one D# per 128x32 tile.
//   * Attention K tiles: global_load_async_to_lds_b128 (ASYNCcnt).
// ---------------------------------------------------------------------------

#define N_HEADS   32
#define N_KV      8
#define HEAD_DIM  128
#define BATCH     2
#define SEQ       1024
#define HID       4096
#define NTOK      (BATCH * SEQ)          // 2048
#define QKV_COLS  (N_HEADS*HEAD_DIM + 2*N_KV*HEAD_DIM)  // 6144

typedef __attribute__((ext_vector_type(16))) __bf16 v16bf;
typedef __attribute__((ext_vector_type(8)))  float  v8f;
typedef unsigned int u32x4 __attribute__((ext_vector_type(4)));
typedef int          i32x4 __attribute__((ext_vector_type(4)));
typedef int          i32x8 __attribute__((ext_vector_type(8)));

// Low 32 bits of a generic pointer into LDS == LDS byte offset (aperture is
// carried in addr[63:32]).
__device__ __forceinline__ unsigned lds_off(const void* p) {
  return (unsigned)(uintptr_t)p;
}

// ---- Tensor Data Mover: 2D tile (tile_rows x 32 bf16) global -> LDS -------
// D# per CDNA5 ISA. Group2/3 NULL-equivalent (2D).
__device__ __forceinline__ void tdm_load_tile_2d(unsigned lds_addr,
                                                 const void* gaddr,
                                                 unsigned k_elems,
                                                 unsigned tile_rows) {
  const unsigned long long ga = (unsigned long long)(uintptr_t)gaddr;
  u32x4 g0;
  g0.x = 1u;                                            // count=1 (valid D#)
  g0.y = lds_addr;                                      // lds_addr [63:32]
  g0.z = (unsigned)ga;                                  // global_addr [95:64]
  g0.w = ((unsigned)(ga >> 32) & 0x01FFFFFFu)           // global_addr [120:96]
         | 0x80000000u;                                 // type=2 ("image")
  i32x8 g1;
  g1[0] = 0x00010000;                    // wg_mask=0 (no cluster), data_size=2B
  g1[1] = (int)(k_elems << 16);          // tensor_dim0[15:0]  (bits 63:48)
  g1[2] = (int)((k_elems >> 16) | (tile_rows << 16));   // dim0 hi | dim1 lo
  g1[3] = (int)(32u << 16);              // tile_dim0 = 32 elems (bits 127:112)
  g1[4] = (int)tile_rows;                // tile_dim1 (bits 143:128)
  g1[5] = (int)k_elems;                  // tensor_dim0_stride[31:0] (elems)
  g1[6] = 0;                             // stride hi | tensor_dim1_stride lo
  g1[7] = 0;
  i32x4 z4 = {};
#if __clang_major__ >= 23
  i32x8 z8 = {};
  __builtin_amdgcn_tensor_load_to_lds(g0, g1, z4, z4, z8, 0);
#else
  __builtin_amdgcn_tensor_load_to_lds(g0, g1, z4, z4, 0);
#endif
}
__device__ __forceinline__ void wait_tensor_all()   {
  __builtin_amdgcn_s_wait_tensorcnt(0);
}
__device__ __forceinline__ void wait_tensor_prev2() {
  __builtin_amdgcn_s_wait_tensorcnt(2);
}

// ---- Async global -> LDS copy, 16 bytes per lane (ASYNCcnt) ---------------
__device__ __forceinline__ void async_copy_b128(unsigned lds_byte_addr,
                                                const void* gaddr) {
  asm volatile("global_load_async_to_lds_b128 %0, %1, off"
               :: "v"(lds_byte_addr), "v"(gaddr) : "memory");
}
__device__ __forceinline__ void wait_async_all() {
  asm volatile("s_wait_asynccnt 0x0" ::: "memory");
}

// Load a 16x32 (bf16) WMMA operand fragment.
// Per ISA 7.12.2, lane L (row/col = L%16, hi = L/16) holds K-runs
// {hi*8 .. hi*8+7} and {16+hi*8 .. 16+hi*8+7} -> exactly two b128 loads
// from a K-contiguous row. rowptr = &src[row*ld + kchunk]; 16B aligned.
__device__ __forceinline__ v16bf load_frag16(const __bf16* rowptr) {
  const int hi = (threadIdx.x & 31) >> 4;
  union { uint4 u[2]; v16bf f; } U;
  U.u[0] = *(const uint4*)(rowptr + hi * 8);
  U.u[1] = *(const uint4*)(rowptr + 16 + hi * 8);
  return U.f;
}

// -------------------------- conversion kernels -----------------------------

__global__ void k_cvt_bf16(const float* __restrict__ src,
                           __bf16* __restrict__ dst, int n) {
  for (int i = blockIdx.x * blockDim.x + threadIdx.x; i < n;
       i += gridDim.x * blockDim.x)
    dst[i] = (__bf16)src[i];
}

// src is (K,N) row-major f32 ; dst is (N,K) row-major bf16 (pre-transposed
// weights so GEMM B-operand rows are K-contiguous).
__global__ void k_cvt_bf16_T(const float* __restrict__ src,
                             __bf16* __restrict__ dst, int K, int N) {
  const int total = K * N;
  for (int i = blockIdx.x * blockDim.x + threadIdx.x; i < total;
       i += gridDim.x * blockDim.x) {
    const int k = i / N, nn = i - k * N;
    dst[(size_t)nn * K + k] = (__bf16)src[i];
  }
}

// ------------------------------- GEMM --------------------------------------
// C(M,N) f32 = A(M,K)bf16 * Bt(N,K)bf16^T.
// 128x128 workgroup tile, 8 waves as 4(row) x 2(col); each wave owns a 32x64
// C region -> 8 WMMAs per K-step from 2 A-frags + 4 B-frags.
// Tiles DMA'd by the TDM (one descriptor per 128x32 tile), double-buffered.

__global__ __launch_bounds__(256) void k_gemm_bf16(
    const __bf16* __restrict__ A, const __bf16* __restrict__ Bt,
    float* __restrict__ C, int M, int N, int K, int ldc) {
  __shared__ __attribute__((aligned(16))) __bf16 ldsA[2][128 * 32];
  __shared__ __attribute__((aligned(16))) __bf16 ldsB[2][128 * 32];
  (void)M; (void)N;
  const int mbase = blockIdx.y * 128, nbase = blockIdx.x * 128;
  const int tid = threadIdx.x, lane = tid & 31, n = lane & 15, hi = lane >> 4;
  const int w = tid >> 5;
  const int rg = (w & 3) * 32;        // wave row base (two 16-row blocks)
  const int cg = (w >> 2) * 64;       // wave col base (four 16-col blocks)

  v8f acc[2][4] = {};

  const __bf16* Atile = A  + (size_t)mbase * K;
  const __bf16* Btile = Bt + (size_t)nbase * K;
  const unsigned la0 = lds_off(&ldsA[0][0]);
  const unsigned lb0 = lds_off(&ldsB[0][0]);
  const unsigned bufstride = 128u * 32u * 2u;   // bytes per LDS buffer

  auto stage = [&](int kt, int b) {   // wave0 only: 2 TDM ops per K-step
    tdm_load_tile_2d(la0 + (unsigned)b * bufstride, Atile + (size_t)kt * 32,
                     (unsigned)K, 128u);
    tdm_load_tile_2d(lb0 + (unsigned)b * bufstride, Btile + (size_t)kt * 32,
                     (unsigned)K, 128u);
  };

  const int nk = K / 32;
  if (w == 0) stage(0, 0);
  for (int kt = 0; kt < nk; ++kt) {
    const int b = kt & 1;
    if (w == 0) {
      if (kt + 1 < nk) {
        stage(kt + 1, b ^ 1);   // stream next tile while computing this one
        wait_tensor_prev2();    // in-order TENSORcnt: this tile's 2 ops done
      } else {
        wait_tensor_all();
      }
    }
    __syncthreads();
    const __bf16* LA = &ldsA[b][0];
    const __bf16* LB = &ldsB[b][0];
    v16bf fa[2], fb[4];
    fa[0] = load_frag16(LA + (rg + n) * 32);
    fa[1] = load_frag16(LA + (rg + 16 + n) * 32);
#pragma unroll
    for (int j = 0; j < 4; ++j)
      fb[j] = load_frag16(LB + (cg + j * 16 + n) * 32);
#pragma unroll
    for (int i = 0; i < 2; ++i)
#pragma unroll
      for (int j = 0; j < 4; ++j)
        acc[i][j] = __builtin_amdgcn_wmma_f32_16x16x32_bf16(
            false, fa[i], false, fb[j], (short)0, acc[i][j], false, false);
    __syncthreads();
  }
#pragma unroll
  for (int i = 0; i < 2; ++i) {
    float* Cw = C + (size_t)(mbase + rg + i * 16 + 8 * hi) * ldc + nbase + cg;
#pragma unroll
    for (int v = 0; v < 8; ++v)
#pragma unroll
      for (int j = 0; j < 4; ++j)
        Cw[(size_t)v * ldc + j * 16 + n] = acc[i][j][v];
  }
}

// --------------------------- RoPE + GQA split ------------------------------
// QKV f32 (NTOK, 6144) -> Qb (NTOK,32,128) bf16, Kb/Vb (NTOK,8,128) bf16.

__global__ __launch_bounds__(128) void k_rope_split(
    const float* __restrict__ QKV, const float* __restrict__ fc,
    const float* __restrict__ fs, __bf16* __restrict__ Qb,
    __bf16* __restrict__ Kb, __bf16* __restrict__ Vb) {
  const int t = blockIdx.x, s = t & (SEQ - 1), tid = threadIdx.x;
  const int j = tid & 63, half = tid >> 6;      // rotary pair index / half
  const float c  = fc[s * 64 + j];
  const float sn = fs[s * 64 + j];
  const float* row = QKV + (size_t)t * QKV_COLS;
#pragma unroll
  for (int h = 0; h < N_HEADS; ++h) {
    const float a = row[h * HEAD_DIM + j], b2 = row[h * HEAD_DIM + 64 + j];
    const float r = half ? (a * sn + b2 * c) : (a * c - b2 * sn);
    Qb[((size_t)t * N_HEADS + h) * HEAD_DIM + tid] = (__bf16)r;
  }
#pragma unroll
  for (int h = 0; h < N_KV; ++h) {
    const float* kr = row + N_HEADS * HEAD_DIM + h * HEAD_DIM;
    const float a = kr[j], b2 = kr[64 + j];
    const float r = half ? (a * sn + b2 * c) : (a * c - b2 * sn);
    Kb[((size_t)t * N_KV + h) * HEAD_DIM + tid] = (__bf16)r;
    Vb[((size_t)t * N_KV + h) * HEAD_DIM + tid] =
        (__bf16)row[(N_HEADS + N_KV) * HEAD_DIM + h * HEAD_DIM + tid];
  }
}

// ---------------------------- flash attention ------------------------------
// Block = (b, h, 64 q rows); 4 waves x 16 q rows. KV in blocks of 32 staged
// in LDS (K async row-major kv x d, V transposed d x kv). Online softmax.

__global__ __launch_bounds__(128) void k_flash_attn(
    const __bf16* __restrict__ Qb, const __bf16* __restrict__ Kb,
    const __bf16* __restrict__ Vb, __bf16* __restrict__ Ob) {
  __shared__ __attribute__((aligned(16))) __bf16 ldsK[32 * HEAD_DIM];
  __shared__ __attribute__((aligned(16))) __bf16 ldsVt[HEAD_DIM * 32];
  __shared__ __attribute__((aligned(16))) __bf16 ldsP[4 * 16 * 32];

  const int bh = blockIdx.x;                 // b*32 + h
  const int b = bh >> 5, h = bh & 31, hk = h >> 2;   // GQA: 4 q heads / kv head
  const int qblk = blockIdx.y;               // 64-row q block
  const int tid = threadIdx.x, lane = tid & 31, n = lane & 15, hi = lane >> 4;
  const int w = tid >> 5;
  const int qrow = qblk * 64 + w * 16;       // wave's q base (seq coord)
  const float scale = 0.08838834764831845f;  // 1/sqrt(128)
  const unsigned ldsKoff = lds_off(ldsK);

  // Q fragments (A operand): 4 K-chunks over HEAD_DIM=128
  const __bf16* qptr =
      Qb + (((size_t)(b * SEQ + qrow + n)) * N_HEADS + h) * HEAD_DIM;
  v16bf fq[4];
#pragma unroll
  for (int kc = 0; kc < 4; ++kc) fq[kc] = load_frag16(qptr + kc * 32);

  v8f acc[8] = {};                           // 16 x 128 f32 output tile
  float mrow[8], lrow[8];
#pragma unroll
  for (int i = 0; i < 8; ++i) { mrow[i] = -1e30f; lrow[i] = 0.f; }

  const int kb_end = 2 * qblk + 1;           // causal bound for this block
  const int wave_qmax = qrow + 15;

  for (int kb = 0; kb <= kb_end; ++kb) {
    const int ks = kb * 32;
    // stage K (32 x 128) via async copies; V loaded + transposed via regs
#pragma unroll
    for (int r = 0; r < 4; ++r) {
      const int idx = tid + r * 128;         // 0..511
      const int row = idx >> 4, seg = idx & 15;
      const size_t kvoff =
          (((size_t)(b * SEQ + ks + row)) * N_KV + hk) * HEAD_DIM + seg * 8;
      async_copy_b128(ldsKoff + (unsigned)idx * 16u, Kb + kvoff);
      union { uint4 u; __bf16 e[8]; } V;
      V.u = *(const uint4*)(Vb + kvoff);
#pragma unroll
      for (int e = 0; e < 8; ++e) ldsVt[(seg * 8 + e) * 32 + row] = V.e[e];
    }
    wait_async_all();
    __syncthreads();

    if (ks <= wave_qmax) {                   // wave-uniform causal skip
      // S = Q K^T : two 16x16 score tiles over 32 kv columns
      v8f s0 = {}; v8f s1 = {};
#pragma unroll
      for (int kc = 0; kc < 4; ++kc) {
        v16bf fk0 = load_frag16(ldsK + (size_t)n * HEAD_DIM + kc * 32);
        v16bf fk1 = load_frag16(ldsK + (size_t)(16 + n) * HEAD_DIM + kc * 32);
        s0 = __builtin_amdgcn_wmma_f32_16x16x32_bf16(false, fq[kc], false, fk0,
                                                     (short)0, s0, false, false);
        s1 = __builtin_amdgcn_wmma_f32_16x16x32_bf16(false, fq[kc], false, fk1,
                                                     (short)0, s1, false, false);
      }
      // online softmax per C row (row = v + 8*hi, 16 lanes = 16 columns)
      float alpha[8];
#pragma unroll
      for (int v = 0; v < 8; ++v) {
        const int q = qrow + v + 8 * hi;
        float a0 = (ks + n      <= q) ? s0[v] * scale : -1e30f;
        float a1 = (ks + 16 + n <= q) ? s1[v] * scale : -1e30f;
        float mloc = fmaxf(a0, a1);
#pragma unroll
        for (int msk = 1; msk <= 8; msk <<= 1)
          mloc = fmaxf(mloc, __shfl_xor(mloc, msk, 32));
        const float mnew = fmaxf(mrow[v], mloc);
        alpha[v] = __expf(mrow[v] - mnew);
        const float p0 = __expf(a0 - mnew), p1 = __expf(a1 - mnew);
        float rs = p0 + p1;
#pragma unroll
        for (int msk = 1; msk <= 8; msk <<= 1) rs += __shfl_xor(rs, msk, 32);
        lrow[v] = lrow[v] * alpha[v] + rs;
        mrow[v] = mnew;
        s0[v] = p0; s1[v] = p1;              // reuse score regs as P
      }
#pragma unroll
      for (int tI = 0; tI < 8; ++tI)
#pragma unroll
        for (int v = 0; v < 8; ++v) acc[tI][v] *= alpha[v];

      // restage P (C layout -> A layout) through per-wave LDS tile
      __bf16* P = ldsP + w * (16 * 32);
#pragma unroll
      for (int v = 0; v < 8; ++v) {
        const int row = v + 8 * hi;
        P[row * 32 + n]      = (__bf16)s0[v];
        P[row * 32 + 16 + n] = (__bf16)s1[v];
      }
      v16bf fp = load_frag16(P + n * 32);    // wave-internal DS: in order
      // O += P * V  (8 output column chunks of 16)
#pragma unroll
      for (int nc = 0; nc < 8; ++nc) {
        v16bf fv = load_frag16(ldsVt + (nc * 16 + n) * 32);
        acc[nc] = __builtin_amdgcn_wmma_f32_16x16x32_bf16(
            false, fp, false, fv, (short)0, acc[nc], false, false);
      }
    }
    __syncthreads();
  }

  // normalize and emit bf16 attention output (token-major, col = h*128+d)
#pragma unroll
  for (int v = 0; v < 8; ++v) {
    const float inv = 1.0f / lrow[v];
    const size_t tok = (size_t)b * SEQ + qrow + v + 8 * hi;
    __bf16* orow = Ob + tok * (N_HEADS * HEAD_DIM) + h * HEAD_DIM;
#pragma unroll
    for (int nc = 0; nc < 8; ++nc)
      orow[nc * 16 + n] = (__bf16)(acc[nc][v] * inv);
  }
}

// ------------------------------- launcher ----------------------------------

extern "C" void kernel_launch(void* const* d_in, const int* in_sizes, int n_in,
                              void* d_out, int out_size, void* d_ws,
                              size_t ws_size, hipStream_t stream) {
  (void)in_sizes; (void)n_in; (void)out_size; (void)ws_size;
  const float* x  = (const float*)d_in[0];
  const float* wq = (const float*)d_in[1];
  const float* wk = (const float*)d_in[2];
  const float* wv = (const float*)d_in[3];
  const float* wo = (const float*)d_in[4];
  const float* fc = (const float*)d_in[5];
  const float* fs = (const float*)d_in[6];
  // d_in[7] = start_pos (reference fixes it to 0)
  float* out = (float*)d_out;

  char* ws = (char*)d_ws;
  size_t off = 0;
  auto alloc = [&](size_t bytes) {
    char* p = ws + off;
    off += (bytes + 255) & ~size_t(255);
    return p;
  };
  __bf16* Xb  = (__bf16*)alloc((size_t)NTOK * HID * 2);        // 16 MB
  __bf16* WqT = (__bf16*)alloc((size_t)4096 * HID * 2);        // 32 MB
  __bf16* WkT = (__bf16*)alloc((size_t)1024 * HID * 2);        //  8 MB
  __bf16* WvT = (__bf16*)alloc((size_t)1024 * HID * 2);        //  8 MB
  __bf16* WoT = (__bf16*)alloc((size_t)HID * 4096 * 2);        // 32 MB
  float*  QKV = (float*) alloc((size_t)NTOK * QKV_COLS * 4);   // 48 MB
  __bf16* Qb  = (__bf16*)alloc((size_t)NTOK * 4096 * 2);       // 16 MB
  __bf16* Kb  = (__bf16*)alloc((size_t)NTOK * 1024 * 2);       //  4 MB
  __bf16* Vb  = (__bf16*)alloc((size_t)NTOK * 1024 * 2);       //  4 MB
  __bf16* Ab  = (__bf16*)alloc((size_t)NTOK * 4096 * 2);       // 16 MB

  // precision prep: bf16 activations, pre-transposed bf16 weights
  k_cvt_bf16  <<<2048, 256, 0, stream>>>(x,  Xb,  NTOK * HID);
  k_cvt_bf16_T<<<2048, 256, 0, stream>>>(wq, WqT, HID, 4096);
  k_cvt_bf16_T<<<2048, 256, 0, stream>>>(wk, WkT, HID, 1024);
  k_cvt_bf16_T<<<2048, 256, 0, stream>>>(wv, WvT, HID, 1024);
  k_cvt_bf16_T<<<2048, 256, 0, stream>>>(wo, WoT, 4096, HID);

  // fused QKV projection into one (NTOK, 6144) f32 buffer
  k_gemm_bf16<<<dim3(32, 16), 256, 0, stream>>>(Xb, WqT, QKV + 0,
                                                NTOK, 4096, HID, QKV_COLS);
  k_gemm_bf16<<<dim3(8, 16), 256, 0, stream>>>(Xb, WkT, QKV + 4096,
                                               NTOK, 1024, HID, QKV_COLS);
  k_gemm_bf16<<<dim3(8, 16), 256, 0, stream>>>(Xb, WvT, QKV + 5120,
                                               NTOK, 1024, HID, QKV_COLS);

  k_rope_split<<<NTOK, 128, 0, stream>>>(QKV, fc, fs, Qb, Kb, Vb);

  k_flash_attn<<<dim3(BATCH * N_HEADS, SEQ / 64), 128, 0, stream>>>(Qb, Kb, Vb,
                                                                    Ab);

  // output projection
  k_gemm_bf16<<<dim3(32, 16), 256, 0, stream>>>(Ab, WoT, out,
                                                NTOK, HID, 4096, HID);
}
